// ExpertAttentionHead_38027640439249
// MI455X (gfx1250) — matrix operand order repaired and checked by reading, not compiled, not run on hardware
//
#include <hip/hip_runtime.h>
#include <math.h>
#include <stdint.h>

// ---------------------------------------------------------------------------
// MI455X (gfx1250) implementation: everything through v_wmma_f32_16x16x32_bf16.
// Reynolds term = row-constant logit shift -> cancels in softmax; only the
// 0.5x temperature survives (SC = 0.5/sqrt(HD)).
// bf16 A-tiles are staged with GLOBAL_LOAD_ASYNC_TO_LDS_B128 (ASYNCcnt path).
// ---------------------------------------------------------------------------

typedef __attribute__((ext_vector_type(16))) __bf16          v16bf;
typedef __attribute__((ext_vector_type(8)))  float           v8f;
typedef __attribute__((ext_vector_type(8)))  unsigned short  u16x8;
typedef __attribute__((ext_vector_type(16))) unsigned short  u16x16;

#define B_  2
#define L_  2048
#define D_  2048
#define H_  16
#define HD_ 128
#define E_  8192
#define M_  (B_*L_)          // 4096 rows for all GEMMs

// float -> bf16 (round to nearest even, ignoring NaN edge case)
static __device__ __forceinline__ unsigned short f2bf(float f){
  unsigned u = __builtin_bit_cast(unsigned, f);
  unsigned r = u + 0x7FFFu + ((u >> 16) & 1u);
  return (unsigned short)(r >> 16);
}

static __device__ __forceinline__ v16bf mkfrag(u16x8 lo, u16x8 hi){
  u16x16 t = __builtin_shufflevector(lo, hi, 0,1,2,3,4,5,6,7,8,9,10,11,12,13,14,15);
  return __builtin_bit_cast(v16bf, t);
}

// load 16 contiguous elements from global memory as bf16 ushorts
static __device__ __forceinline__ void ld16(const float* g, unsigned short* dst){
  const float4* p = (const float4*)g;
  #pragma unroll
  for (int i = 0; i < 4; ++i){
    float4 v = p[i];
    dst[i*4+0] = f2bf(v.x); dst[i*4+1] = f2bf(v.y);
    dst[i*4+2] = f2bf(v.z); dst[i*4+3] = f2bf(v.w);
  }
}
static __device__ __forceinline__ void ld16(const unsigned short* g, unsigned short* dst){
  const u16x8* p = (const u16x8*)g;
  u16x8 a = p[0], b = p[1];
  #pragma unroll
  for (int i = 0; i < 8; ++i){ dst[i] = a[i]; dst[8+i] = b[i]; }
}

static __device__ __forceinline__ void stC(float* p, float v){ *p = v; }
static __device__ __forceinline__ void stC(unsigned short* p, float v){ *p = f2bf(v); }

// ---- A-tile staging: 16 contiguous bf16 elements global -> LDS -------------
// fp32 source: convert through registers. bf16 source: CDNA5 async copy
// (GLOBAL_LOAD_ASYNC_TO_LDS_B128); the instruction IOFFSET advances both the
// LDS and global addresses, so offset:0/16 stage 32 bytes per lane.
static __device__ __forceinline__ void stageA(const float* g, unsigned short* dst){
  unsigned short tmp[16];
  ld16(g, tmp);
  *(u16x8*)dst       = *(const u16x8*)&tmp[0];
  *(u16x8*)(dst + 8) = *(const u16x8*)&tmp[8];
}
static __device__ __forceinline__ void stageA(const unsigned short* g, unsigned short* dst){
  unsigned lds            = (unsigned)(uintptr_t)dst;        // low 32 bits = LDS offset
  unsigned long long gadr = (unsigned long long)(uintptr_t)g;
  asm volatile("global_load_async_to_lds_b128 %0, %1, off\n\t"
               "global_load_async_to_lds_b128 %0, %1, off offset:16"
               :: "v"(lds), "v"(gadr) : "memory");
}
static __device__ __forceinline__ void wait_async(){
  asm volatile("s_wait_asynccnt 0x0" ::: "memory");
}

// ---------------------------------------------------------------------------
// Generic bf16-WMMA GEMM: C(M,N) = A(M,K) * B(K,N), row-major.
// Block tile 128x128, BK=32, 256 threads = 8 waves; wave tile 32x64.
// LDS row stride 40 ushorts (80B) -> conflict-free b128 frag reads.
// ---------------------------------------------------------------------------
template<class AT, class CT>
__global__ __launch_bounds__(256)
void gemm_wmma(const AT* __restrict__ A, const float* __restrict__ Bw,
               CT* __restrict__ C, int M, int N, int K){
  __shared__ unsigned short As[128*40];   // [m][k], k contiguous
  __shared__ unsigned short Bs[128*40];   // [n][k], k contiguous (transposed stage)
  const int t    = threadIdx.x;
  const int lane = t & 31, w = t >> 5;
  const int g    = lane >> 4, l15 = lane & 15;
  const int wr   = w >> 1, wc = w & 1;     // 4x2 wave grid: 32 rows x 64 cols
  const int m0   = blockIdx.y * 128, n0 = blockIdx.x * 128;

  v8f zero = {0.f,0.f,0.f,0.f,0.f,0.f,0.f,0.f};
  v8f acc[2][4];
  #pragma unroll
  for (int i = 0; i < 2; ++i)
    #pragma unroll
    for (int j = 0; j < 4; ++j) acc[i][j] = zero;

  const int ar = t >> 1, ac = (t & 1) * 16;   // A stage: 128 rows x 32 k
  const int bk = t >> 3, bn = (t & 7) * 16;   // B stage: 32 k x 128 n

  for (int k0 = 0; k0 < K; k0 += 32){
    __syncthreads();
    {
      stageA(&A[(size_t)(m0 + ar) * K + k0 + ac], &As[ar*40 + ac]);
      unsigned short tmp[16];
      ld16(&Bw[(size_t)(k0 + bk) * N + n0 + bn], tmp);
      #pragma unroll
      for (int i = 0; i < 16; ++i) Bs[(bn + i)*40 + bk] = tmp[i];
      if (k0 + 32 < K){
        __builtin_prefetch(&A[(size_t)(m0 + ar) * K + k0 + 32 + ac], 0, 1);
        __builtin_prefetch(&Bw[(size_t)(k0 + 32 + bk) * N + n0 + bn], 0, 1);
      }
    }
    wait_async();                // async A-tile copies complete before barrier
    __syncthreads();

    v16bf af[2], bfr[4];
    #pragma unroll
    for (int mt = 0; mt < 2; ++mt){
      int mloc = wr*32 + mt*16 + l15;
      u16x8 lo = *(const u16x8*)&As[mloc*40 + g*8];        // k = 8g..8g+7
      u16x8 hi = *(const u16x8*)&As[mloc*40 + g*8 + 16];   // k = 16+8g..
      af[mt] = mkfrag(lo, hi);
    }
    #pragma unroll
    for (int nt = 0; nt < 4; ++nt){
      int nloc = wc*64 + nt*16 + l15;
      u16x8 lo = *(const u16x8*)&Bs[nloc*40 + g*8];        // k = 8g..8g+7
      u16x8 hi = *(const u16x8*)&Bs[nloc*40 + g*8 + 16];   // k = 16+8g..
      bfr[nt] = mkfrag(lo, hi);
    }
    #pragma unroll
    for (int mt = 0; mt < 2; ++mt)
      #pragma unroll
      for (int nt = 0; nt < 4; ++nt)
        acc[mt][nt] = __builtin_amdgcn_wmma_f32_16x16x32_bf16(
            false, af[mt], false, bfr[nt], (short)0, acc[mt][nt], false, false);
  }

  #pragma unroll
  for (int mt = 0; mt < 2; ++mt)
    #pragma unroll
    for (int nt = 0; nt < 4; ++nt)
      #pragma unroll
      for (int vr = 0; vr < 8; ++vr){
        int row = m0 + wr*32 + mt*16 + vr + 8*g;   // C layout: lanes16-31 -> M+8
        int col = n0 + wc*64 + nt*16 + l15;
        stC(&C[(size_t)row * N + col], acc[mt][nt][vr]);
      }
}

// ---------------------------------------------------------------------------
// Fused dual GEMM + SwiGLU: H = silu(A*B1) * (A*B2), A bf16, B f32, H bf16.
// Block tile 128x64, 8 waves, each wave owns one 16-row stripe x 64 cols.
// A staged via async-to-LDS (bf16 straight copy).
// ---------------------------------------------------------------------------
__global__ __launch_bounds__(256)
void gemm_dual_swiglu(const unsigned short* __restrict__ A,
                      const float* __restrict__ B1, const float* __restrict__ B2,
                      unsigned short* __restrict__ Hout, int M, int N, int K){
  __shared__ unsigned short As [128*40];
  __shared__ unsigned short B1s[64*40];
  __shared__ unsigned short B2s[64*40];
  const int t    = threadIdx.x;
  const int lane = t & 31, w = t >> 5;
  const int g    = lane >> 4, l15 = lane & 15;
  const int m0   = blockIdx.y * 128, n0 = blockIdx.x * 64;

  v8f zero = {0.f,0.f,0.f,0.f,0.f,0.f,0.f,0.f};
  v8f a1[4], a2[4];
  #pragma unroll
  for (int i = 0; i < 4; ++i){ a1[i] = zero; a2[i] = zero; }

  const int ar = t >> 1,  ac = (t & 1) * 16;
  const int ts = t & 127;
  const int bk = ts >> 2, bn = (ts & 3) * 16;   // 32 k-rows x 64 n, 128 threads/matrix

  for (int k0 = 0; k0 < K; k0 += 32){
    __syncthreads();
    {
      stageA(&A[(size_t)(m0 + ar) * K + k0 + ac], &As[ar*40 + ac]);
      unsigned short tmp[16];
      const float* src = (t < 128) ? B1 : B2;
      unsigned short* dst = (t < 128) ? B1s : B2s;
      ld16(&src[(size_t)(k0 + bk) * N + n0 + bn], tmp);
      #pragma unroll
      for (int i = 0; i < 16; ++i) dst[(bn + i)*40 + bk] = tmp[i];
    }
    wait_async();
    __syncthreads();

    v16bf af;
    {
      int mloc = w*16 + l15;
      u16x8 lo = *(const u16x8*)&As[mloc*40 + g*8];
      u16x8 hi = *(const u16x8*)&As[mloc*40 + g*8 + 16];
      af = mkfrag(lo, hi);
    }
    v16bf f1[4], f2[4];
    #pragma unroll
    for (int nt = 0; nt < 4; ++nt){
      int nloc = nt*16 + l15;
      f1[nt] = mkfrag(*(const u16x8*)&B1s[nloc*40 + g*8],
                      *(const u16x8*)&B1s[nloc*40 + g*8 + 16]);
      f2[nt] = mkfrag(*(const u16x8*)&B2s[nloc*40 + g*8],
                      *(const u16x8*)&B2s[nloc*40 + g*8 + 16]);
    }
    #pragma unroll
    for (int nt = 0; nt < 4; ++nt){
      a1[nt] = __builtin_amdgcn_wmma_f32_16x16x32_bf16(false, af, false, f1[nt], (short)0, a1[nt], false, false);
      a2[nt] = __builtin_amdgcn_wmma_f32_16x16x32_bf16(false, af, false, f2[nt], (short)0, a2[nt], false, false);
    }
  }

  #pragma unroll
  for (int nt = 0; nt < 4; ++nt)
    #pragma unroll
    for (int vr = 0; vr < 8; ++vr){
      int row = m0 + w*16 + vr + 8*g;
      int col = n0 + nt*16 + l15;
      float x1 = a1[nt][vr], x2 = a2[nt][vr];
      float h  = (x1 / (1.f + __expf(-x1))) * x2;    // silu(x1) * x2
      Hout[(size_t)row * N + col] = f2bf(h);
    }
}

// ---------------------------------------------------------------------------
// Causal flash attention with WMMA (both QK^T and P*V).
// Block = 8 waves = 128 query rows of one (b,h); wave owns 16 rows.
// Logit scale 0.5/sqrt(HD): the Reynolds row-mean cancels under softmax.
// Q/K frags loaded directly from global in WMMA layout (contiguous in hd).
// V staged transposed in LDS; P routed via per-wave LDS buffer (same-wave DS
// ops are processed in order, so no block barrier is needed for P).
// K and V frags are preloaded in batches so the 8-WMMA chains issue
// back-to-back behind a single wait.
// ---------------------------------------------------------------------------
__global__ __launch_bounds__(256)
void attn_wmma(const unsigned short* __restrict__ Q,
               const unsigned short* __restrict__ Kd,
               const unsigned short* __restrict__ V,
               unsigned short* __restrict__ O){
  __shared__ unsigned short Vlds[128*40];     // [hd][key] transposed, 10 KB
  __shared__ unsigned short Pbuf[8*16*40];    // per-wave 16x32 P tiles, 10 KB
  const int t    = threadIdx.x;
  const int lane = t & 31, w = t >> 5;
  const int g    = lane >> 4, l15 = lane & 15;
  const int bid  = blockIdx.x;
  const int s = bid & 15, h = (bid >> 4) & 15, b = bid >> 8;
  const int m0 = s*128 + w*16;                 // this wave's first query row
  const size_t base = ((size_t)b * L_) * D_ + (size_t)h * HD_;
  const unsigned short* q = Q  + base;
  const unsigned short* k = Kd + base;
  const unsigned short* v = V  + base;

  // Preload Q A-frags for the 4 hd-chunks of K=128
  v16bf qf[4];
  #pragma unroll
  for (int c = 0; c < 4; ++c){
    const unsigned short* p = q + (size_t)(m0 + l15) * D_ + c*32 + g*8;
    u16x8 lo = *(const u16x8*)p;
    u16x8 hi = *(const u16x8*)(p + 16);
    qf[c] = mkfrag(lo, hi);
  }

  v8f zero = {0.f,0.f,0.f,0.f,0.f,0.f,0.f,0.f};
  float mrow[8], lrow[8];
  v8f acc[8];                                  // O accumulator: 16 rows x 128 hd
  #pragma unroll
  for (int i = 0; i < 8; ++i){ mrow[i] = -3.0e38f; lrow[i] = 0.f; acc[i] = zero; }

  unsigned short* pb = &Pbuf[w * 640];
  const float SC = 0.044194173824159216f;      // 0.5 / sqrt(128)

  const int vkv = t >> 3, vhs = (t & 7) * 16;  // V staging: 32 keys x 128 hd
  const int kend = s*4 + 4;                    // key blocks needed for this stripe
  for (int kb = 0; kb < kend; ++kb){
    __syncthreads();
    {   // stage V block transposed into LDS (shared by all 8 waves)
      unsigned short tmp[16];
      ld16(v + (size_t)(kb*32 + vkv) * D_ + vhs, tmp);
      #pragma unroll
      for (int i = 0; i < 16; ++i) Vlds[(vhs + i)*40 + vkv] = tmp[i];
    }
    __syncthreads();

    if (kb*32 <= m0 + 15){                     // wave-uniform causal skip
      // ---- scores S = Q K^T (16 x 32): batch-load all 8 K frags ----
      v16bf kf[4][2];
      #pragma unroll
      for (int c = 0; c < 4; ++c)
        #pragma unroll
        for (int kt = 0; kt < 2; ++kt){
          const unsigned short* p = k + (size_t)(kb*32 + kt*16 + l15) * D_ + c*32 + g*8;
          kf[c][kt] = mkfrag(*(const u16x8*)p, *(const u16x8*)(p + 16));
        }
      v8f sc[2]; sc[0] = zero; sc[1] = zero;
      #pragma unroll
      for (int c = 0; c < 4; ++c)
        #pragma unroll
        for (int kt = 0; kt < 2; ++kt)
          sc[kt] = __builtin_amdgcn_wmma_f32_16x16x32_bf16(
              false, qf[c], false, kf[c][kt], (short)0, sc[kt], false, false);

      // ---- causal mask + online softmax (row = vr + 8g in C layout) ----
      #pragma unroll
      for (int vr = 0; vr < 8; ++vr){
        int row = m0 + vr + 8*g;
        float s0 = sc[0][vr] * SC, s1 = sc[1][vr] * SC;
        int c0 = kb*32 + l15, c1 = c0 + 16;
        if (c0 > row) s0 = -3.0e38f;
        if (c1 > row) s1 = -3.0e38f;
        float ml = fmaxf(s0, s1);
        #pragma unroll
        for (int mk = 1; mk < 16; mk <<= 1) ml = fmaxf(ml, __shfl_xor(ml, mk, 32));
        float mnew = fmaxf(mrow[vr], ml);
        float p0 = __expf(s0 - mnew), p1 = __expf(s1 - mnew);
        float ps = p0 + p1;
        #pragma unroll
        for (int mk = 1; mk < 16; mk <<= 1) ps += __shfl_xor(ps, mk, 32);
        float alpha = __expf(mrow[vr] - mnew);
        lrow[vr] = lrow[vr]*alpha + ps;
        mrow[vr] = mnew;
        #pragma unroll
        for (int vt = 0; vt < 8; ++vt) acc[vt][vr] *= alpha;
        pb[(vr + 8*g)*40 + l15]      = f2bf(p0);
        pb[(vr + 8*g)*40 + 16 + l15] = f2bf(p1);
      }
      asm volatile("" ::: "memory");   // pin DS store->load order (in-order per wave)

      // ---- O += P V: batch-load P frag + all 8 V frags, then 8 WMMAs ----
      {
        const unsigned short* pp = &pb[l15*40 + g*8];
        v16bf pf = mkfrag(*(const u16x8*)pp, *(const u16x8*)(pp + 16));
        v16bf vf[8];
        #pragma unroll
        for (int vt = 0; vt < 8; ++vt){
          const unsigned short* vp = &Vlds[(vt*16 + l15)*40 + g*8];
          vf[vt] = mkfrag(*(const u16x8*)vp, *(const u16x8*)(vp + 16));
        }
        #pragma unroll
        for (int vt = 0; vt < 8; ++vt)
          acc[vt] = __builtin_amdgcn_wmma_f32_16x16x32_bf16(
              false, pf, false, vf[vt], (short)0, acc[vt], false, false);
      }
    }
  }

  // normalize and write bf16 output at (b, row, h*128 + hd)
  #pragma unroll
  for (int vr = 0; vr < 8; ++vr){
    float rinv = 1.0f / lrow[vr];
    int row = m0 + vr + 8*g;
    #pragma unroll
    for (int vt = 0; vt < 8; ++vt)
      O[((size_t)b * L_ + row) * D_ + h*HD_ + vt*16 + l15] = f2bf(acc[vt][vr] * rinv);
  }
}

// ---------------------------------------------------------------------------
// Host-side pipeline (all on `stream`, graph-capture safe).
// Inputs: x, mask(unused; causal mask applied analytically), Wq, Wk, Wv, W1, Vg, W2
// ---------------------------------------------------------------------------
extern "C" void kernel_launch(void* const* d_in, const int* in_sizes, int n_in,
                              void* d_out, int out_size, void* d_ws, size_t ws_size,
                              hipStream_t stream){
  (void)in_sizes; (void)n_in; (void)out_size; (void)ws_size;
  const float* x  = (const float*)d_in[0];
  const float* Wq = (const float*)d_in[2];
  const float* Wk = (const float*)d_in[3];
  const float* Wv = (const float*)d_in[4];
  const float* W1 = (const float*)d_in[5];
  const float* Vg = (const float*)d_in[6];
  const float* W2 = (const float*)d_in[7];
  float* out = (float*)d_out;

  // bf16 workspace: q,k,v,attn (M*D each) + h (M*E)  ->  ~134 MB
  unsigned short* qb = (unsigned short*)d_ws;
  unsigned short* kb = qb + (size_t)M_ * D_;
  unsigned short* vb = kb + (size_t)M_ * D_;
  unsigned short* ab = vb + (size_t)M_ * D_;
  unsigned short* hb = ab + (size_t)M_ * D_;

  dim3 blk(256);
  dim3 gproj(D_/128, M_/128);
  hipLaunchKernelGGL((gemm_wmma<float, unsigned short>), gproj, blk, 0, stream,
                     x, Wq, qb, M_, D_, D_);
  hipLaunchKernelGGL((gemm_wmma<float, unsigned short>), gproj, blk, 0, stream,
                     x, Wk, kb, M_, D_, D_);
  hipLaunchKernelGGL((gemm_wmma<float, unsigned short>), gproj, blk, 0, stream,
                     x, Wv, vb, M_, D_, D_);

  hipLaunchKernelGGL(attn_wmma, dim3(B_ * H_ * (L_/128)), blk, 0, stream,
                     qb, kb, vb, ab);

  hipLaunchKernelGGL(gemm_dual_swiglu, dim3(E_/64, M_/128), blk, 0, stream,
                     ab, W1, Vg, hb, M_, E_, D_);

  hipLaunchKernelGGL((gemm_wmma<unsigned short, float>), dim3(D_/128, M_/128), blk, 0, stream,
                     hb, W2, out, M_, D_, E_);
}
